// multi_head_attention_25340307046910
// MI455X (gfx1250) — compile-verified, matrix operand order checked by reading, not run on hardware
//
#include <hip/hip_runtime.h>

// ---------------------------------------------------------------------------
// MI455X (gfx1250) multi-head attention forward.
// wave32, WMMA 16x16x32 bf16, 320KB-LDS score stripes, async LDS staging.
// ---------------------------------------------------------------------------

typedef __attribute__((ext_vector_type(16))) __bf16          v16bf;
typedef __attribute__((ext_vector_type(16))) unsigned short  v16u;
typedef __attribute__((ext_vector_type(8)))  float           v8f;

#define Bn   2
#define Sn   2048
#define Dn   1024
#define Hn   16
#define DHn  64
#define Mn   (Bn * Sn)      // 4096 rows in all projection GEMMs

__device__ __forceinline__ unsigned short f32_to_bf16(float f) {
  unsigned int u = __builtin_bit_cast(unsigned int, f);
  u += 0x7FFFu + ((u >> 16) & 1u);          // round-to-nearest-even
  return (unsigned short)(u >> 16);
}

// K index of half-pair p (p=0..7) for a lane, per ISA 16-bit A/B layout:
// lanes 0-15 hold K {0..7,16..23}, lanes 16-31 hold K {8..15,24..31}.
__device__ __forceinline__ int kpair_base(int p, int lane) {
  return 2 * p + ((p >= 4) ? 8 : 0) + ((lane >= 16) ? 8 : 0);
}

__device__ __forceinline__ v8f zero8() {
  v8f z;
  for (int i = 0; i < 8; ++i) z[i] = 0.0f;
  return z;
}

__device__ __forceinline__ v8f wmma_bf16(v16u a, v16u b, v8f c) {
  return __builtin_amdgcn_wmma_f32_16x16x32_bf16(
      /*neg_a=*/false, __builtin_bit_cast(v16bf, a),
      /*neg_b=*/false, __builtin_bit_cast(v16bf, b),
      /*c_mod=*/(short)0, c, /*reuse_a=*/false, /*reuse_b=*/false);
}

// Async copy of one 16B chunk: global -> LDS (ASYNCcnt-tracked, CDNA5 ISA).
__device__ __forceinline__ void async_copy_b128(unsigned lds_off,
                                                const void* gaddr) {
  asm volatile("global_load_async_to_lds_b128 %0, %1, off"
               :: "v"(lds_off), "v"((unsigned long long)(size_t)gaddr)
               : "memory");
}
__device__ __forceinline__ void wait_asynccnt0() {
  asm volatile("s_wait_asynccnt 0x0" ::: "memory");
}

// ---------------------------------------------------------------------------
// GEMM:  C[4096 x 1024] = X[4096 x 1024] * W[1024 x 1024] + bias
// mode 0: out = bf16, head-split [B,H,S,DH] layout (QKV projections)
// mode 1: out = fp32, plain [B,S,D] layout (output projection)
// Block: 128 threads (4 waves), 64x64 block tile, K-step 32.
// ---------------------------------------------------------------------------
#define LDA 40   // padded LDS row stride (bf16 elems, 80B -> 16B aligned)

__global__ __launch_bounds__(128)
void gemm_bias_kernel(const float* __restrict__ X, const float* __restrict__ W,
                      const float* __restrict__ bias, void* __restrict__ out,
                      int mode) {
  __shared__ unsigned short As[64 * LDA];   // A tile  [m][k]
  __shared__ unsigned short Bs[64 * LDA];   // B tile transposed: [n][k]

  const int tid  = threadIdx.x;
  const int lane = tid & 31;
  const int wave = tid >> 5;
  const int m0   = (blockIdx.x >> 4) << 6;  // 64 row-blocks
  const int n0   = (blockIdx.x & 15) << 6;  // 16 col-blocks

  v8f acc[4];
  for (int i = 0; i < 4; ++i) acc[i] = zero8();

  for (int k0 = 0; k0 < Dn; k0 += 32) {
    if (k0 + 32 < Dn)
      __builtin_prefetch(X + (m0 + (tid >> 3)) * Dn + k0 + 32, 0, 1);

    // Stage A: 64x32 fp32 -> bf16  (512 float4 chunks / 128 threads)
    for (int i = tid; i < 512; i += 128) {
      const int r  = i >> 3;           // row 0..63
      const int c4 = (i & 7) << 2;     // col 0..28 step 4
      const float4 f = *(const float4*)(X + (m0 + r) * Dn + k0 + c4);
      unsigned short* dst = &As[r * LDA + c4];
      dst[0] = f32_to_bf16(f.x); dst[1] = f32_to_bf16(f.y);
      dst[2] = f32_to_bf16(f.z); dst[3] = f32_to_bf16(f.w);
    }
    // Stage B transposed: W[k0+kk][n0+n] -> Bs[n][kk]
    for (int i = tid; i < 512; i += 128) {
      const int kk = i >> 4;           // 0..31
      const int c4 = (i & 15) << 2;    // n 0..60 step 4
      const float4 f = *(const float4*)(W + (k0 + kk) * Dn + n0 + c4);
      Bs[(c4 + 0) * LDA + kk] = f32_to_bf16(f.x);
      Bs[(c4 + 1) * LDA + kk] = f32_to_bf16(f.y);
      Bs[(c4 + 2) * LDA + kk] = f32_to_bf16(f.z);
      Bs[(c4 + 3) * LDA + kk] = f32_to_bf16(f.w);
    }
    __syncthreads();

    // A fragment for this wave's 16 rows
    const int arow = (wave << 4) + (lane & 15);
    v16u a;
    for (int p = 0; p < 8; ++p) {
      const int kb = kpair_base(p, lane);
      const unsigned int u = *(const unsigned int*)&As[arow * LDA + kb];
      a[2 * p]     = (unsigned short)u;
      a[2 * p + 1] = (unsigned short)(u >> 16);
    }
    for (int nt = 0; nt < 4; ++nt) {
      const int bcol = (nt << 4) + (lane & 15);
      v16u b;
      for (int p = 0; p < 8; ++p) {
        const int kb = kpair_base(p, lane);
        const unsigned int u = *(const unsigned int*)&Bs[bcol * LDA + kb];
        b[2 * p]     = (unsigned short)u;
        b[2 * p + 1] = (unsigned short)(u >> 16);
      }
      acc[nt] = wmma_bf16(a, b, acc[nt]);
    }
    __syncthreads();
  }

  // Epilogue: C/D layout -> lane l, vgpr r holds (M = r + 8*(l>=16), N = l%16)
  for (int nt = 0; nt < 4; ++nt) {
    for (int r = 0; r < 8; ++r) {
      const int mglob = m0 + (wave << 4) + r + ((lane >> 4) << 3);
      const int nglob = n0 + (nt << 4) + (lane & 15);
      const float val = acc[nt][r] + bias[nglob];
      if (mode == 0) {
        const int b = mglob >> 11, srow = mglob & (Sn - 1);
        const int h = nglob >> 6,  dh   = nglob & (DHn - 1);
        ((unsigned short*)out)[(((b * Hn + h) * Sn) + srow) * DHn + dh] =
            f32_to_bf16(val);
      } else {
        ((float*)out)[(size_t)mglob * Dn + nglob] = val;
      }
    }
  }
}

// ---------------------------------------------------------------------------
// Attention core: one block per (b, h, 16-query-row stripe).
// 128 threads (4 waves). Full 16x2048 fp32 score stripe lives in LDS (128KB,
// possible thanks to CDNA5's 320KB/WGP). soft_mat written exactly once.
// V k-tiles staged wave-privately via global_load_async_to_lds_b128.
// ---------------------------------------------------------------------------
#define SS_FLOATS (16 * Sn)          // 32768 fp32   = 131072 B
#define SQ_STRIDE 72                 // padded bf16 row stride (144B)
#define SQ_BYTES  (16 * SQ_STRIDE * 2)
#define SRED_OFF  (SS_FLOATS * 4 + SQ_BYTES)
#define SV_OFF    (SRED_OFF + 256 * 4)
#define SV_TILE_B 4096               // 32 x 64 bf16 per wave
#define ATTN_SMEM (SV_OFF + 4 * SV_TILE_B)

__global__ __launch_bounds__(128)
void attn_kernel(const unsigned short* __restrict__ qh,
                 const unsigned short* __restrict__ kh,
                 const unsigned short* __restrict__ vh,
                 float* __restrict__ soft, float* __restrict__ concat) {
  extern __shared__ char smem_raw[];
  float*          sS   = (float*)smem_raw;                       // scores stripe
  unsigned short* sQ   = (unsigned short*)(smem_raw + SS_FLOATS * 4);
  float*          sRed = (float*)(smem_raw + SRED_OFF);

  const int tid  = threadIdx.x;
  const int lane = tid & 31;
  const int wave = tid >> 5;
  const int bh   = blockIdx.x >> 7;          // b*H + h   (0..31)
  const int q0   = (blockIdx.x & 127) << 4;  // query row base

  // Wave-private V staging tile (32 x 64 bf16)
  unsigned short* sV = (unsigned short*)(smem_raw + SV_OFF + wave * SV_TILE_B);
  const unsigned  sV_lds = (unsigned)(size_t)(void*)sV;  // LDS aperture offset

  // ---- Stage Q tile: 16 x 64 bf16 ----
  {
    const unsigned short* qbase = qh + ((size_t)bh * Sn + q0) * DHn;
    const int r = tid >> 3, c8 = (tid & 7) << 3;
    *(uint4*)&sQ[r * SQ_STRIDE + c8] = *(const uint4*)(qbase + r * DHn + c8);
  }
  __syncthreads();

  // A fragments for q (K = DH = 64 -> two k-steps of 32)
  v16u aq[2];
  {
    const int arow = lane & 15;
    for (int ks = 0; ks < 2; ++ks)
      for (int p = 0; p < 8; ++p) {
        const int kb = ks * 32 + kpair_base(p, lane);
        const unsigned int u = *(const unsigned int*)&sQ[arow * SQ_STRIDE + kb];
        aq[ks][2 * p]     = (unsigned short)u;
        aq[ks][2 * p + 1] = (unsigned short)(u >> 16);
      }
  }

  // ---- Scores: wave w covers key columns [w*512, w*512+512) ----
  const unsigned short* kbase = kh + (size_t)bh * Sn * DHn;
  for (int ct = wave * 32; ct < wave * 32 + 32; ++ct) {
    const int c0 = ct << 4;
    v8f s = zero8();
    for (int ks = 0; ks < 2; ++ks) {
      v16u bfrag;
      const int col = c0 + (lane & 15);
      for (int p = 0; p < 8; ++p) {
        const int kb = ks * 32 + kpair_base(p, lane);
        const unsigned int u = *(const unsigned int*)(kbase + col * DHn + kb);
        bfrag[2 * p]     = (unsigned short)u;
        bfrag[2 * p + 1] = (unsigned short)(u >> 16);
      }
      s = wmma_bf16(aq[ks], bfrag, s);
    }
    for (int r = 0; r < 8; ++r) {
      const int m = r + ((lane >> 4) << 3);
      sS[m * Sn + c0 + (lane & 15)] = s[r] * (1.0f / 4096.0f);  // /DH^2
    }
  }
  __syncthreads();

  // ---- Softmax over 2048 keys per row (8 threads per row) ----
  const int row = tid >> 3, sub = tid & 7;
  float mx = -1e30f;
  for (int c = sub; c < Sn; c += 8) mx = fmaxf(mx, sS[row * Sn + c]);
  sRed[tid] = mx;
  __syncthreads();
  if (sub == 0) {
    float m2 = sRed[tid];
    for (int i = 1; i < 8; ++i) m2 = fmaxf(m2, sRed[tid + i]);
    sRed[128 + row] = m2;
  }
  __syncthreads();
  const float rowmax = sRed[128 + row];
  float sum = 0.0f;
  for (int c = sub; c < Sn; c += 8) {
    const float e = __expf(sS[row * Sn + c] - rowmax);
    sS[row * Sn + c] = e;
    sum += e;
  }
  __syncthreads();
  sRed[tid] = sum;
  __syncthreads();
  if (sub == 0) {
    float s2 = 0.0f;
    for (int i = 0; i < 8; ++i) s2 += sRed[tid + i];
    sRed[128 + row] = s2;
  }
  __syncthreads();
  const float inv = 1.0f / sRed[128 + row];

  // Normalize in LDS and stream soft_mat out (single write of the 537MB blob)
  float* softrow = soft + ((size_t)bh * Sn + q0) * Sn;
  for (int c = sub; c < Sn; c += 8) {
    const float p = sS[row * Sn + c] * inv;
    sS[row * Sn + c] = p;
    softrow[(size_t)row * Sn + c] = p;
  }
  __syncthreads();

  // ---- attn = P @ V ; wave w reduces its 512-key slice ----
  const unsigned short* vbase = vh + (size_t)bh * Sn * DHn;
  v8f oacc[4];
  for (int i = 0; i < 4; ++i) oacc[i] = zero8();

  for (int ks = 0; ks < 16; ++ks) {
    const int kglob = wave * 512 + ks * 32;

    // Async-stage this wave's V k-tile (32 x 64 bf16 = 4KB) into LDS.
    // Wave-private region: ASYNCcnt wait is sufficient, no barrier needed.
    {
      const char* gsrc = (const char*)(vbase + (size_t)kglob * DHn);
      for (int j = 0; j < 8; ++j) {
        const unsigned chunk = (unsigned)(j * 32 + lane) * 16u;  // 16B chunks
        async_copy_b128(sV_lds + chunk, gsrc + chunk);
      }
      wait_asynccnt0();
    }

    v16u ap;  // P fragment, fp32 LDS -> bf16 on the fly
    {
      const int m = lane & 15;
      for (int p = 0; p < 8; ++p) {
        const int kb = kglob + kpair_base(p, lane);
        const float2 f = *(const float2*)&sS[m * Sn + kb];
        ap[2 * p]     = f32_to_bf16(f.x);
        ap[2 * p + 1] = f32_to_bf16(f.y);
      }
    }
    for (int nt = 0; nt < 4; ++nt) {
      v16u bv;
      const int n = (nt << 4) + (lane & 15);
      for (int h2 = 0; h2 < 16; ++h2) {
        const int kk = h2 + ((h2 >= 8) ? 8 : 0) + ((lane >= 16) ? 8 : 0);
        bv[h2] = sV[kk * DHn + n];           // gather from staged LDS tile
      }
      oacc[nt] = wmma_bf16(ap, bv, oacc[nt]);
    }
  }
  __syncthreads();                 // all waves done reading P from sS

  // Cross-wave reduction through LDS (reuse score stripe region)
  float* part = sS;                // [wave][16][64]
  for (int nt = 0; nt < 4; ++nt)
    for (int r = 0; r < 8; ++r) {
      const int m = r + ((lane >> 4) << 3);
      part[wave * 1024 + m * DHn + (nt << 4) + (lane & 15)] = oacc[nt][r];
    }
  __syncthreads();

  const int b = bh >> 4, h = bh & 15;
  for (int i = tid; i < 1024; i += 128) {
    const int m = i >> 6, n = i & 63;
    const float v = part[m * DHn + n] + part[1024 + m * DHn + n] +
                    part[2048 + m * DHn + n] + part[3072 + m * DHn + n];
    concat[((size_t)(b * Sn) + q0 + m) * Dn + h * DHn + n] = v;
  }
}

// ---------------------------------------------------------------------------
extern "C" void kernel_launch(void* const* d_in, const int* in_sizes, int n_in,
                              void* d_out, int out_size, void* d_ws, size_t ws_size,
                              hipStream_t stream) {
  (void)in_sizes; (void)n_in; (void)out_size; (void)ws_size;

  const float* Q  = (const float*)d_in[0];
  const float* K  = (const float*)d_in[1];
  const float* V  = (const float*)d_in[2];
  const float* Wq = (const float*)d_in[3];
  const float* bq = (const float*)d_in[4];
  const float* Wk = (const float*)d_in[5];
  const float* bk = (const float*)d_in[6];
  const float* Wv = (const float*)d_in[7];
  const float* bv = (const float*)d_in[8];
  const float* Wo = (const float*)d_in[9];
  const float* bo = (const float*)d_in[10];

  // Workspace: qh/kh/vh bf16 [B,H,S,DH] (8MB each) + concat fp32 [B,S,D] (16MB)
  unsigned short* qh = (unsigned short*)d_ws;
  unsigned short* kh = qh + (size_t)Bn * Hn * Sn * DHn;
  unsigned short* vh = kh + (size_t)Bn * Hn * Sn * DHn;
  float* concat = (float*)(vh + (size_t)Bn * Hn * Sn * DHn);

  float* outp = (float*)d_out;                          // [B,S,D] fp32
  float* soft = outp + (size_t)Bn * Sn * Dn;            // [B,H,S,S] fp32

  const dim3 gemm_grid((Mn / 64) * (Dn / 64));          // 1024 blocks
  const dim3 blk(128);

  gemm_bias_kernel<<<gemm_grid, blk, 0, stream>>>(Q, Wq, bq, qh, 0);
  gemm_bias_kernel<<<gemm_grid, blk, 0, stream>>>(K, Wk, bk, kh, 0);
  gemm_bias_kernel<<<gemm_grid, blk, 0, stream>>>(V, Wv, bv, vh, 0);

  attn_kernel<<<dim3(Bn * Hn * (Sn / 16)), blk, ATTN_SMEM, stream>>>(
      qh, kh, vh, soft, concat);

  gemm_bias_kernel<<<gemm_grid, blk, 0, stream>>>(concat, Wo, bo, outp, 1);
}
